// residual_LSTM_30691836297313
// MI455X (gfx1250) — matrix operand order repaired and checked by reading, not compiled, not run on hardware
//
#include <hip/hip_runtime.h>
#include <hip/hip_bf16.h>

typedef __attribute__((ext_vector_type(16))) __bf16 v16bf;
typedef __attribute__((ext_vector_type(8)))  float  v8f;

#define NB   32      // batch
#define NT   1024    // time
#define NH   512     // hidden
#define NL   2       // layers
#define NG   2048    // 4*H
#define NWG  64      // persistent workgroups
#define NTHR 128     // threads per WG (4 waves)

// LDS weight panel layout (elements): per wave, per layer, 16 rows of K=1024
// padded to 1032 so row stride = 2064B (mod 256 = 16) -> bank-conflict-free.
#define ROWE 1032
#define LAYE (16 * ROWE)          // 16512 elements per layer panel
#define WAVE_E (NL * LAYE)        // 33024 elements per wave
#define SMEM_BYTES (4 * WAVE_E * 2)  // 264192 bytes (<= 320KB WGP LDS)

// ---- workspace layout (bytes), all regions 256B aligned ----
static constexpr size_t SYNC_OFF = 0;                                   // 2 uints + pad
static constexpr size_t Z_OFF    = 256;                                 // f32 [32][2048]
static constexpr size_t C_OFF    = Z_OFF + (size_t)NB * NG * 4;         // f32 [2][32][512]
static constexpr size_t HB_OFF   = C_OFF + (size_t)NL * NB * NH * 4;    // bf16 [2][32][512]
static constexpr size_t WI_OFF   = HB_OFF + (size_t)NL * NB * NH * 2;   // bf16 [2][2048][512]
static constexpr size_t WS_OFF   = WI_OFF + (size_t)NL * NG * NH * 2;   // bf16 [2][2048][512]
static constexpr size_t XB_OFF   = WS_OFF + (size_t)NL * NG * NH * 2;   // bf16 [32][1024][512]

__global__ void lstm_init_kernel(const float* __restrict__ x,
                                 const float* __restrict__ Wi,
                                 const float* __restrict__ Ws,
                                 char* __restrict__ ws) {
    unsigned*        sync = (unsigned*)(ws + SYNC_OFF);
    float*           C    = (float*)(ws + C_OFF);
    __hip_bfloat16*  Hb   = (__hip_bfloat16*)(ws + HB_OFF);
    __hip_bfloat16*  Wib  = (__hip_bfloat16*)(ws + WI_OFF);
    __hip_bfloat16*  Wsb  = (__hip_bfloat16*)(ws + WS_OFF);
    __hip_bfloat16*  Xb   = (__hip_bfloat16*)(ws + XB_OFF);

    const size_t NX = (size_t)NB * NT * NH;          // 16,777,216
    const size_t NW = (size_t)NL * NG * NH;          //  2,097,152
    const size_t NS = (size_t)NL * NB * NH;          //     32,768

    size_t i      = (size_t)blockIdx.x * blockDim.x + threadIdx.x;
    size_t stride = (size_t)gridDim.x * blockDim.x;
    for (size_t k = i; k < NX; k += stride) {
        Xb[k] = __float2bfloat16(x[k]);
        if (k < NW) { Wib[k] = __float2bfloat16(Wi[k]); Wsb[k] = __float2bfloat16(Ws[k]); }
        if (k < NS) { C[k] = 0.0f; Hb[k] = __float2bfloat16(0.0f); }
        if (k < 64)  sync[k] = 0u;
    }
}

__device__ __forceinline__ void grid_barrier(unsigned* counter, unsigned* gen) {
    __syncthreads();
    if (threadIdx.x == 0) {
        __threadfence();
        unsigned g = __hip_atomic_load(gen, __ATOMIC_ACQUIRE, __HIP_MEMORY_SCOPE_AGENT);
        unsigned arrived = __hip_atomic_fetch_add(counter, 1u, __ATOMIC_ACQ_REL,
                                                  __HIP_MEMORY_SCOPE_AGENT);
        if (arrived == NWG - 1) {
            __hip_atomic_store(counter, 0u, __ATOMIC_RELAXED, __HIP_MEMORY_SCOPE_AGENT);
            __hip_atomic_fetch_add(gen, 1u, __ATOMIC_ACQ_REL, __HIP_MEMORY_SCOPE_AGENT);
        } else {
            while (__hip_atomic_load(gen, __ATOMIC_ACQUIRE, __HIP_MEMORY_SCOPE_AGENT) == g) {
                __builtin_amdgcn_s_sleep(1);
            }
        }
        __threadfence();
    }
    __syncthreads();
}

__device__ __forceinline__ float sigmoidf_(float v) { return 1.0f / (1.0f + __expf(-v)); }

__global__ __launch_bounds__(NTHR, 1)
void lstm_persistent_kernel(const float* __restrict__ x,
                            const float* __restrict__ bi,
                            const float* __restrict__ bs,
                            float* __restrict__ out,
                            char* __restrict__ ws) {
    extern __shared__ __align__(16) char smem[];
    __hip_bfloat16* sW = (__hip_bfloat16*)smem;

    unsigned*              sync = (unsigned*)(ws + SYNC_OFF);
    float*                 Z    = (float*)(ws + Z_OFF);
    float*                 C    = (float*)(ws + C_OFF);
    __hip_bfloat16*        Hb   = (__hip_bfloat16*)(ws + HB_OFF);
    const __hip_bfloat16*  Wib  = (const __hip_bfloat16*)(ws + WI_OFF);
    const __hip_bfloat16*  Wsb  = (const __hip_bfloat16*)(ws + WS_OFF);
    const __hip_bfloat16*  Xb   = (const __hip_bfloat16*)(ws + XB_OFF);

    const int lane = threadIdx.x & 31;
    const int wave = threadIdx.x >> 5;
    const int gw   = blockIdx.x * 4 + wave;       // 0..255 wave -> one 16x16 tile
    const int mt   = gw >> 7;                      // 0..1   (M tile: batch rows)
    const int nt   = gw & 127;                     // 0..127 (N tile: output cols)

    const int rowA = mt * 16 + (lane & 15);        // batch row this lane loads (A frag)
    const int hiA  = (lane >> 4) * 8;              // A: lanes 16-31 hold K+8 chunk
    const int hiB  = (lane >> 4) * 8;              // B: lanes 16-31 hold K+8 chunk

    const int tidg = blockIdx.x * NTHR + threadIdx.x;  // 0..8191

    // ---- stage this wave's weight panels into LDS (reused 1024 timesteps) -------
    // Panel k-axis = [Wi row (k<512) | Ws row (k>=512)], matching the kb loop below.
    for (int j = 0; j < NL; ++j) {
        for (int r = 0; r < 16; ++r) {
            __hip_bfloat16*       dst   = sW + wave * WAVE_E + j * LAYE + r * ROWE;
            const __hip_bfloat16* srcWi = Wib + ((size_t)j * NG + nt * 16 + r) * NH;
            const __hip_bfloat16* srcWs = Wsb + ((size_t)j * NG + nt * 16 + r) * NH;
            ((uint4*)dst)[lane]             = ((const uint4*)srcWi)[lane];
            ((uint4*)dst)[lane + 32]        = ((const uint4*)srcWi)[lane + 32];
            ((uint4*)(dst + NH))[lane]      = ((const uint4*)srcWs)[lane];
            ((uint4*)(dst + NH))[lane + 32] = ((const uint4*)srcWs)[lane + 32];
        }
    }
    __syncthreads();

    for (int t = 0; t < NT; ++t) {
        for (int j = 0; j < NL; ++j) {
            // ---------------- GEMM phase: z = [inp | h] @ [Wi | Ws]^T --------------
            v8f acc = {};
            const __hip_bfloat16* inpRow = (j == 0)
                ? (Xb + ((size_t)rowA * NT + t) * NH)            // layer0 input = x_t
                : (Hb + (size_t)rowA * NH);                      // layer1 input = h0
            const __hip_bfloat16* hRow = Hb + (size_t)j * NB * NH + (size_t)rowA * NH;
            const __hip_bfloat16* sWj  = sW + wave * WAVE_E + j * LAYE
                                            + (lane & 15) * ROWE;

            #pragma unroll 4
            for (int kb = 0; kb < 32; ++kb) {
                const __hip_bfloat16* ap;
                int kl;
                if (kb < 16) { ap = inpRow; kl = kb * 32; }
                else         { ap = hRow;   kl = (kb - 16) * 32; }

                union { uint4 u[2]; v16bf v; } Af, Bf;
                Af.u[0] = *(const uint4*)(ap + kl + hiA);        // K chunk {0..7}/{8..15}
                Af.u[1] = *(const uint4*)(ap + kl + hiA + 16);   // K chunk +16

                const __hip_bfloat16* bp = sWj + kb * 32 + hiB;  // LDS weight run
                Bf.u[0] = *(const uint4*)(bp);
                Bf.u[1] = *(const uint4*)(bp + 16);

                acc = __builtin_amdgcn_wmma_f32_16x16x32_bf16(
                          false, Af.v, false, Bf.v, (short)0, acc, false, false);
            }
            // store 16x16 fp32 tile: VGPR g -> row (mt*16 + g + 8*(lane>=16)), col n*16+(lane&15)
            {
                const int mrow = mt * 16 + (lane >> 4) * 8;
                const int ncol = nt * 16 + (lane & 15);
                #pragma unroll
                for (int g = 0; g < 8; ++g)
                    Z[(size_t)(mrow + g) * NG + ncol] = acc[g];
            }
            grid_barrier(sync, sync + 1);

            // ---------------- Elementwise gate phase ------------------------------
            for (int e = tidg; e < NB * NH; e += NWG * NTHR) {
                const int b  = e >> 9;
                const int ch = e & 511;
                const float* zrow = Z + (size_t)b * NG;
                const size_t bj   = (size_t)j * NG;

                float zi = zrow[ch]            + bi[bj + ch]            + bs[bj + ch];
                float zf = zrow[NH + ch]       + bi[bj + NH + ch]       + bs[bj + NH + ch];
                float zg = zrow[2 * NH + ch]   + bi[bj + 2 * NH + ch]   + bs[bj + 2 * NH + ch];
                float zo = zrow[3 * NH + ch]   + bi[bj + 3 * NH + ch]   + bs[bj + 3 * NH + ch];

                float ig = sigmoidf_(zi);
                float fg = sigmoidf_(zf);
                float gg = tanhf(zg);
                float og = sigmoidf_(zo);

                const size_t si = (size_t)j * NB * NH + e;
                float cn = ig * gg + fg * C[si];
                C[si] = cn;

                float inpv = (j == 0)
                    ? x[((size_t)b * NT + t) * NH + ch]
                    : __bfloat162float(Hb[e]);                   // h0 (layer-1 residual)
                float hn = og * tanhf(cn) + inpv;
                Hb[si] = __float2bfloat16(hn);
                if (j == 1)
                    out[((size_t)b * NT + t) * NH + ch] = hn;
            }
            grid_barrier(sync, sync + 1);
        }
    }

    // ---------------- final states: h_f, c_f as [B, L, H] -------------------------
    float* hf = out + (size_t)NB * NT * NH;
    float* cf = hf + (size_t)NB * NL * NH;
    for (int e = tidg; e < NL * NB * NH; e += NWG * NTHR) {
        const int j  = e / (NB * NH);
        const int r  = e - j * (NB * NH);
        const int b  = r >> 9;
        const int ch = r & 511;
        const size_t oidx = ((size_t)b * NL + j) * NH + ch;
        hf[oidx] = __bfloat162float(Hb[e]);
        cf[oidx] = C[e];
    }
}

extern "C" void kernel_launch(void* const* d_in, const int* in_sizes, int n_in,
                              void* d_out, int out_size, void* d_ws, size_t ws_size,
                              hipStream_t stream) {
    const float* x  = (const float*)d_in[0];
    const float* Wi = (const float*)d_in[1];
    const float* bi = (const float*)d_in[2];
    const float* Ws = (const float*)d_in[3];
    const float* bs = (const float*)d_in[4];
    float* out = (float*)d_out;
    char*  ws  = (char*)d_ws;

    (void)in_sizes; (void)n_in; (void)out_size; (void)ws_size;

    // allow >64KB dynamic LDS (264KB of the WGP's 320KB); host-side, capture-safe
    hipFuncSetAttribute((const void*)lstm_persistent_kernel,
                        hipFuncAttributeMaxDynamicSharedMemorySize, SMEM_BYTES);

    lstm_init_kernel<<<2048, 256, 0, stream>>>(x, Wi, Ws, ws);
    lstm_persistent_kernel<<<NWG, NTHR, SMEM_BYTES, stream>>>(x, bi, bs, out, ws);
}
